// GRACE_44822278701655
// MI455X (gfx1250) — compile-verified, matrix operand order checked by reading, not run on hardware
//
#include <hip/hip_runtime.h>
#include <hip/hip_bf16.h>
#include <cstdint>

typedef __attribute__((ext_vector_type(16))) __bf16 v16bf;
typedef __attribute__((ext_vector_type(8)))  float  v8f;

#define DIM 256

// ---------------------------------------------------------------------------
// WMMA helper: D = A(16x32 bf16) * B^T-tile(32x16 bf16) + C(16x16 f32)
// ---------------------------------------------------------------------------
__device__ __forceinline__ v8f wmma_bf16(v16bf a, v16bf b, v8f c) {
  return __builtin_amdgcn_wmma_f32_16x16x32_bf16(
      /*neg_a=*/false, a, /*neg_b=*/false, b,
      /*c_mod=*/(short)0, c, /*reuse_a=*/false, /*reuse_b=*/false);
}

// A/B fragment loader for bf16 row-major source (A·B^T form: B frag = rows of
// the second matrix).  CDNA5 16-bit A layout: lane L holds row (L&15); lane
// group g = L>>4 holds K = {g*8 .. g*8+7} in elems 0-7 and K = {16+g*8 ..} in
// elems 8-15.  Two 16-byte loads per lane.
__device__ __forceinline__ v16bf load_frag_bf16(const __bf16* base, int stride,
                                                int rowbase, int k0, int lane) {
  int row = rowbase + (lane & 15);
  int g   = lane >> 4;
  const __bf16* p = base + (size_t)row * stride + k0 + g * 8;
  v16bf f;
  *reinterpret_cast<uint4*>(&f)       = *reinterpret_cast<const uint4*>(p);
  *(reinterpret_cast<uint4*>(&f) + 1) = *reinterpret_cast<const uint4*>(p + 16);
  return f;
}

// Same fragment, but source is fp32 row-major; convert to bf16 on the fly.
__device__ __forceinline__ v16bf load_frag_f32(const float* base, int stride,
                                               int rowbase, int k0, int lane) {
  int row = rowbase + (lane & 15);
  int g   = lane >> 4;
  const float* p = base + (size_t)row * stride + k0 + g * 8;
  float4 q0 = *reinterpret_cast<const float4*>(p);
  float4 q1 = *reinterpret_cast<const float4*>(p + 4);
  float4 q2 = *reinterpret_cast<const float4*>(p + 16);
  float4 q3 = *reinterpret_cast<const float4*>(p + 20);
  v16bf f;
  f[0]=(__bf16)q0.x;  f[1]=(__bf16)q0.y;  f[2]=(__bf16)q0.z;  f[3]=(__bf16)q0.w;
  f[4]=(__bf16)q1.x;  f[5]=(__bf16)q1.y;  f[6]=(__bf16)q1.z;  f[7]=(__bf16)q1.w;
  f[8]=(__bf16)q2.x;  f[9]=(__bf16)q2.y;  f[10]=(__bf16)q2.z; f[11]=(__bf16)q2.w;
  f[12]=(__bf16)q3.x; f[13]=(__bf16)q3.y; f[14]=(__bf16)q3.z; f[15]=(__bf16)q3.w;
  return f;
}

// ---------------------------------------------------------------------------
// Kernel 0: zero accumulators
// ---------------------------------------------------------------------------
__global__ void zero_kernel(float* p, int n) {
  int i = blockIdx.x * blockDim.x + threadIdx.x;
  if (i < n) p[i] = 0.0f;
}

// ---------------------------------------------------------------------------
// Kernel 1: projection + row L2-normalize.
//   h = ELU(z @ W1^T + b1) @ W2^T + b2 ;  n = h / ||h||  (bf16 out)
// 64 threads = 2 waves; each wave owns 16 rows.  A-fragments for each layer
// are hoisted into registers (they do not depend on the output column tile).
// ---------------------------------------------------------------------------
#define H1S 264   // padded bf16 row stride (bank-conflict avoidance, 16B-mult)
#define HBS 260   // padded f32 row stride (16B-mult)

__global__ void __launch_bounds__(64)
project_kernel(const float* __restrict__ z,
               const float* __restrict__ W1, const float* __restrict__ b1,
               const float* __restrict__ W2, const float* __restrict__ b2,
               __bf16* __restrict__ nout) {
  __shared__ __align__(16) __bf16 h1buf[2][16][H1S];
  __shared__ __align__(16) float  hbuf [2][16][HBS];

  const int lane = threadIdx.x & 31;
  const int wv   = threadIdx.x >> 5;
  const int r0   = (blockIdx.x * 2 + wv) * 16;
  const int nc   = lane & 15;
  const int mb   = (lane >> 4) << 3;

  // ---- layer 1: t = z @ W1^T + b1 ; ELU ; stage bf16 in LDS ----
  {
    v16bf af[8];
    #pragma unroll
    for (int kk = 0; kk < 8; ++kk)
      af[kk] = load_frag_f32(z, DIM, r0, kk * 32, lane);

    for (int ct = 0; ct < 16; ++ct) {
      v8f c = {0.f,0.f,0.f,0.f,0.f,0.f,0.f,0.f};
      #pragma unroll
      for (int kk = 0; kk < 8; ++kk)
        c = wmma_bf16(af[kk], load_frag_f32(W1, DIM, ct * 16, kk * 32, lane), c);
      float bias = b1[ct * 16 + nc];
      #pragma unroll
      for (int v = 0; v < 8; ++v) {
        float x = c[v] + bias;
        x = x > 0.0f ? x : (__expf(x) - 1.0f);      // ELU
        h1buf[wv][v + mb][ct * 16 + nc] = (__bf16)x;
      }
    }
  }
  __syncthreads();

  // ---- layer 2: h = elu_h @ W2^T + b2 ; stage f32 in LDS ----
  {
    v16bf af[8];
    #pragma unroll
    for (int kk = 0; kk < 8; ++kk)
      af[kk] = load_frag_bf16(&h1buf[wv][0][0], H1S, 0, kk * 32, lane);

    for (int ct = 0; ct < 16; ++ct) {
      v8f c = {0.f,0.f,0.f,0.f,0.f,0.f,0.f,0.f};
      #pragma unroll
      for (int kk = 0; kk < 8; ++kk)
        c = wmma_bf16(af[kk], load_frag_f32(W2, DIM, ct * 16, kk * 32, lane), c);
      float bias = b2[ct * 16 + nc];
      #pragma unroll
      for (int v = 0; v < 8; ++v)
        hbuf[wv][v + mb][ct * 16 + nc] = c[v] + bias;
    }
  }
  __syncthreads();

  // ---- row L2-normalize, emit bf16 ----
  for (int m = 0; m < 16; ++m) {
    float x[8];
    float ss = 0.0f;
    #pragma unroll
    for (int kk = 0; kk < 8; ++kk) {
      x[kk] = hbuf[wv][m][lane + 32 * kk];
      ss += x[kk] * x[kk];
    }
    #pragma unroll
    for (int off = 1; off < 32; off <<= 1) ss += __shfl_xor(ss, off);
    float inv = rsqrtf(ss);
    #pragma unroll
    for (int kk = 0; kk < 8; ++kk)
      nout[(size_t)(r0 + m) * DIM + lane + 32 * kk] = (__bf16)(x[kk] * inv);
  }
}

// ---------------------------------------------------------------------------
// Tile epilogue: e = exp(2*s); accumulate row sums (+ optional col sums),
// store diagonal.  C-layout: lane holds col N = lane&15; VGPR v holds row
// M = v + 8*(lane>>4).
// ---------------------------------------------------------------------------
__device__ __forceinline__ void reduce_tile(v8f acc, int rg, int cg, int lane,
                                            float* rowsum, float* colsum,
                                            float* diag) {
  float e[8];
  #pragma unroll
  for (int v = 0; v < 8; ++v) e[v] = __expf(2.0f * acc[v]);  // 1/TAU = 2

  const int nc = lane & 15;
  const int mb = (lane >> 4) << 3;

  if (diag && rg == cg) {          // tiles are 16-aligned: diagonal iff rg==cg
    #pragma unroll
    for (int v = 0; v < 8; ++v)
      if (v + mb == nc) diag[rg + nc] = e[v];
  }

  if (colsum) {
    float cs = 0.0f;
    #pragma unroll
    for (int v = 0; v < 8; ++v) cs += e[v];
    cs += __shfl_xor(cs, 16);      // combine M=0..7 with M=8..15 halves
    if (lane < 16) atomicAdd(&colsum[cg + lane], cs);
  }

  // butterfly across the 16 lanes of each half-group -> full row sums
  #pragma unroll
  for (int off = 1; off < 16; off <<= 1) {
    #pragma unroll
    for (int v = 0; v < 8; ++v) e[v] += __shfl_xor(e[v], off);
  }
  if (nc == 0) {
    #pragma unroll
    for (int v = 0; v < 8; ++v) atomicAdd(&rowsum[rg + v + mb], e[v]);
  }
}

// ---------------------------------------------------------------------------
// Kernel 2: the three N x N similarity products + exp + reductions.
// 256 threads = 8 waves (4 row-subblocks x 2 col-subblocks); each wave owns a
// 32x32 block = 2x2 WMMA tiles for each of {between, refl1, refl2}.
// Workgroup tile: 128 rows x 64 cols.
//
// The B-side (column) panels of n1/n2 are shared by all 4 row-waves, so they
// are staged in LDS with CDNA5 async global->LDS copies (ASYNCcnt), one
// 128-wide K slice at a time (~34 KB LDS, 272B padded row stride so the
// 16-lane b128 fragment reads are bank-conflict free).
// ---------------------------------------------------------------------------
#define BPS 136   // bf16 elements per padded LDS panel row (128 data + 8 pad)

__global__ void __launch_bounds__(256)
sim_kernel(const __bf16* __restrict__ n1, const __bf16* __restrict__ n2,
           float* rs_r1, float* rs_r2, float* rs_bt, float* cs_bt,
           float* pos, float* dg1, float* dg2) {
  __shared__ __align__(16) __bf16 bpan[2][64][BPS];   // [matrix][panel row][k]

  const int lane  = threadIdx.x & 31;
  const int w     = threadIdx.x >> 5;
  const int wr    = w & 3;
  const int wc    = w >> 2;
  const int row0  = blockIdx.y * 128 + wr * 32;
  const int colB  = blockIdx.x * 64;       // workgroup column-panel base
  const int col0L = wc * 32;               // wave's column offset inside panel

  const v8f zv = {0.f,0.f,0.f,0.f,0.f,0.f,0.f,0.f};
  v8f bt[2][2], r1[2][2], r2[2][2];
  #pragma unroll
  for (int i = 0; i < 2; ++i)
    #pragma unroll
    for (int j = 0; j < 2; ++j) { bt[i][j] = zv; r1[i][j] = zv; r2[i][j] = zv; }

  for (int kb = 0; kb < DIM; kb += 128) {
    __syncthreads();   // previous slice fully consumed before overwrite

    // --- async-stage this K slice of both column panels into LDS ---
    // 2 matrices * 64 rows * 16 chunks(16B) = 2048 chunks; 256 thr -> 8 each.
    #pragma unroll
    for (int m = 0; m < 2; ++m) {
      const __bf16* src = (m == 0) ? n1 : n2;
      #pragma unroll
      for (int rnd = 0; rnd < 4; ++rnd) {
        int q  = (int)threadIdx.x + rnd * 256;
        int r  = q >> 4;        // panel row 0..63
        int ch = q & 15;        // 16B chunk within the 128-elem slice
        unsigned lds_off = (unsigned)(uintptr_t)&bpan[m][r][ch * 8];
        unsigned long long gaddr =
            (unsigned long long)(uintptr_t)(src + (size_t)(colB + r) * DIM +
                                            kb + ch * 8);
        asm volatile("global_load_async_to_lds_b128 %0, %1, off"
                     :: "v"(lds_off), "v"(gaddr) : "memory");
      }
    }
    asm volatile("s_wait_asynccnt 0x0" ::: "memory");
    __syncthreads();   // panels visible to all waves

    // --- consume the slice: A from global (L2), B from LDS ---
    #pragma unroll
    for (int k0 = 0; k0 < 128; k0 += 32) {
      const int kg = kb + k0;
      v16bf a1[2], a2[2], b1f[2], b2f[2];
      #pragma unroll
      for (int i = 0; i < 2; ++i) {
        a1[i]  = load_frag_bf16(n1, DIM, row0 + 16 * i, kg, lane);
        a2[i]  = load_frag_bf16(n2, DIM, row0 + 16 * i, kg, lane);
        b1f[i] = load_frag_bf16(&bpan[0][0][0], BPS, col0L + 16 * i, k0, lane);
        b2f[i] = load_frag_bf16(&bpan[1][0][0], BPS, col0L + 16 * i, k0, lane);
      }
      #pragma unroll
      for (int i = 0; i < 2; ++i)
        #pragma unroll
        for (int j = 0; j < 2; ++j) {
          bt[i][j] = wmma_bf16(a1[i], b2f[j], bt[i][j]);  // n1 . n2^T
          r1[i][j] = wmma_bf16(a1[i], b1f[j], r1[i][j]);  // n1 . n1^T
          r2[i][j] = wmma_bf16(a2[i], b2f[j], r2[i][j]);  // n2 . n2^T
        }
    }
  }

  #pragma unroll
  for (int i = 0; i < 2; ++i)
    #pragma unroll
    for (int j = 0; j < 2; ++j) {
      int rg = row0 + 16 * i, cg = colB + col0L + 16 * j;
      reduce_tile(bt[i][j], rg, cg, lane, rs_bt, cs_bt, pos);
      reduce_tile(r1[i][j], rg, cg, lane, rs_r1, nullptr, dg1);
      reduce_tile(r2[i][j], rg, cg, lane, rs_r2, nullptr, dg2);
    }
}

// ---------------------------------------------------------------------------
// Kernel 3: per-row losses + mean -> scalar
//   denom1_i = rs_r1 + rs_bt - dg1 ; denom2_i = rs_r2 + cs_bt - dg2
//   loss_i   = 0.5 * (-log(pos/denom1) - log(pos/denom2))
// ---------------------------------------------------------------------------
__global__ void __launch_bounds__(256)
loss_kernel(const float* rs_r1, const float* rs_r2, const float* rs_bt,
            const float* cs_bt, const float* pos, const float* dg1,
            const float* dg2, float* out, int N) {
  __shared__ float sbuf[256];
  const int tid = threadIdx.x;
  float local = 0.0f;
  for (int i = tid; i < N; i += 256) {
    float p  = pos[i];
    float d1 = rs_r1[i] + rs_bt[i] - dg1[i];
    float d2 = rs_r2[i] + cs_bt[i] - dg2[i];
    local += 0.5f * (-logf(p / d1) - logf(p / d2));
  }
  sbuf[tid] = local;
  __syncthreads();
  for (int s = 128; s > 0; s >>= 1) {
    if (tid < s) sbuf[tid] += sbuf[tid + s];
    __syncthreads();
  }
  if (tid == 0) out[0] = sbuf[0] / (float)N;
}

// ---------------------------------------------------------------------------
// Launch
// ---------------------------------------------------------------------------
extern "C" void kernel_launch(void* const* d_in, const int* in_sizes, int n_in,
                              void* d_out, int out_size, void* d_ws,
                              size_t ws_size, hipStream_t stream) {
  (void)n_in; (void)out_size; (void)ws_size;
  const float* z1 = (const float*)d_in[0];
  const float* z2 = (const float*)d_in[1];
  const float* W1 = (const float*)d_in[2];
  const float* b1 = (const float*)d_in[3];
  const float* W2 = (const float*)d_in[4];
  const float* b2 = (const float*)d_in[5];

  const int N = in_sizes[0] / DIM;             // 12288
  const size_t nelem = (size_t)N * DIM;

  // workspace layout: n1 bf16 | n2 bf16 | 7 fp32 accumulator arrays of N
  __bf16* n1 = (__bf16*)d_ws;
  __bf16* n2 = n1 + nelem;
  float* acc  = (float*)((char*)d_ws + 2 * nelem * sizeof(__bf16));
  float* rs_r1 = acc + 0 * (size_t)N;
  float* rs_r2 = acc + 1 * (size_t)N;
  float* rs_bt = acc + 2 * (size_t)N;
  float* cs_bt = acc + 3 * (size_t)N;
  float* pos   = acc + 4 * (size_t)N;
  float* dg1   = acc + 5 * (size_t)N;
  float* dg2   = acc + 6 * (size_t)N;

  const int accN = 7 * N;
  zero_kernel<<<(accN + 255) / 256, 256, 0, stream>>>(acc, accN);

  project_kernel<<<N / 32, 64, 0, stream>>>(z1, W1, b1, W2, b2, n1);
  project_kernel<<<N / 32, 64, 0, stream>>>(z2, W1, b1, W2, b2, n2);

  sim_kernel<<<dim3(N / 64, N / 128), 256, 0, stream>>>(
      n1, n2, rs_r1, rs_r2, rs_bt, cs_bt, pos, dg1, dg2);

  loss_kernel<<<1, 256, 0, stream>>>(rs_r1, rs_r2, rs_bt, cs_bt, pos, dg1, dg2,
                                     (float*)d_out, N);
}
